// CrossMambaformer_12867722019119
// MI455X (gfx1250) — matrix operand order repaired and checked by reading, not compiled, not run on hardware
//
#include <hip/hip_runtime.h>
#include <math.h>

// ---------------------------------------------------------------------------
// CrossMambaformer forward for MI455X (gfx1250, wave32, WMMA).
// All heavy math routed through v_wmma_f32_16x16x32_bf16 (fp32 accum).
// GEMM: 128x128x32 tiles, double-buffered LDS, one barrier per k-step,
// next-tile global loads overlapped with WMMA compute.
// ---------------------------------------------------------------------------

#define kDM   1024          // d_model
#define kDI   2048          // d_inner
#define kDS   16            // d_state
#define kDR   64            // dt_rank
#define kNH   16            // heads
#define kHD   64            // head dim
#define kBB   4             // batch
#define kLL   512           // seq len
#define kM    (kBB * kLL)   // 2048 rows

typedef __attribute__((ext_vector_type(16))) __bf16 bf16x16;
typedef __attribute__((ext_vector_type(8)))  __bf16 bf16x8;
typedef __attribute__((ext_vector_type(4)))  __bf16 bf16x4;
typedef __attribute__((ext_vector_type(8)))  float  f32x8;

__device__ __forceinline__ float fast_sigmoid(float x) {
    return 1.0f / (1.0f + __expf(-x));
}

// ---------------------------------------------------------------------------
// Generic batched/strided GEMM:  C = act(alpha * A x B(^T) + bias) + res
//   A: [M,K] fp32, row stride lda.   REQUIRES: M % 128 == 0, K % 32 == 0.
//   B: bNormal=0 -> weights [N,K] (torch Linear layout), stride ldb
//      bNormal=1 -> matrix  [K,N] (e.g. V in attention),  stride ldb
//   N may be ragged (guarded on B loads and C stores only).
//   batch z -> (zo = z / binner, zi = z % binner); per-operand offsets.
//   act: 0 none, 1 relu, 2 silu, 3 softplus
// ---------------------------------------------------------------------------
__global__ __launch_bounds__(256)
void gemm_bf16_wmma(const float* __restrict__ Ag, int lda, long long aso, long long asi,
                    const float* __restrict__ Bg, int ldb, long long bso, long long bsi,
                    int bNormal,
                    const float* __restrict__ bias,
                    const float* __restrict__ res, int ldr, long long rso, long long rsi,
                    float* __restrict__ Cg, int ldc, long long cso, long long csi,
                    int M, int N, int K, int binner, float alpha, int act)
{
    __shared__ __bf16 sA[2][128][40];   // [buf][m][k], padded stride 40 (16B-aligned rows)
    __shared__ __bf16 sB[2][128][40];   // [buf][n][k]

    const int z  = blockIdx.z;
    const int zo = z / binner, zi = z % binner;
    Ag += zo * aso + zi * asi;
    Bg += zo * bso + zi * bsi;
    Cg += zo * cso + zi * csi;
    if (res) res += zo * rso + zi * rsi;

    const int m0 = blockIdx.y * 128;
    const int n0 = blockIdx.x * 128;

    const int tid  = threadIdx.x;
    const int lane = tid & 31;
    const int wave = tid >> 5;
    const int wm   = (wave >> 1) * 32;   // wave row base within tile
    const int wn   = (wave & 1) * 64;    // wave col base within tile

    const bool nfull = (n0 + 128 <= N);

    // staging maps (per thread)
    const int arow = tid >> 3;           // 0..31 (4 passes of 32 rows)
    const int akc  = (tid & 7) * 4;      // 0..28
    const int bkr  = tid >> 5;           // 0..7  (bNormal: 4 passes of 8 k-rows)
    const int bnc  = (tid & 31) * 4;     // 0..124
    const bool bn_ok = nfull || (n0 + bnc + 3 < N);

    f32x8 acc[2][4];
#pragma unroll
    for (int i = 0; i < 2; ++i)
#pragma unroll
        for (int j = 0; j < 4; ++j)
            acc[i][j] = (f32x8)(0.0f);

    float4 aReg[4], bReg[4];

    // ---- prologue: stage k-tile 0 into registers -----------------------
#pragma unroll
    for (int p = 0; p < 4; ++p)
        aReg[p] = *(const float4*)(Ag + (long long)(m0 + p * 32 + arow) * lda + akc);
    if (!bNormal) {
#pragma unroll
        for (int p = 0; p < 4; ++p) {
            int gn = n0 + p * 32 + arow;
            bReg[p] = (nfull || gn < N)
                    ? *(const float4*)(Bg + (long long)gn * ldb + akc)
                    : make_float4(0.f, 0.f, 0.f, 0.f);
        }
    } else {
#pragma unroll
        for (int p = 0; p < 4; ++p) {
            int gk = p * 8 + bkr;
            bReg[p] = bn_ok
                    ? *(const float4*)(Bg + (long long)gk * ldb + n0 + bnc)
                    : make_float4(0.f, 0.f, 0.f, 0.f);
        }
    }

    int buf = 0;
    for (int k0 = 0; k0 < K; k0 += 32) {
        // ---- commit staged registers into LDS[buf] (fp32 -> bf16) ------
#pragma unroll
        for (int p = 0; p < 4; ++p) {
            bf16x4 h;
            h[0] = (__bf16)aReg[p].x; h[1] = (__bf16)aReg[p].y;
            h[2] = (__bf16)aReg[p].z; h[3] = (__bf16)aReg[p].w;
            *(bf16x4*)&sA[buf][p * 32 + arow][akc] = h;
        }
        if (!bNormal) {
#pragma unroll
            for (int p = 0; p < 4; ++p) {
                bf16x4 h;
                h[0] = (__bf16)bReg[p].x; h[1] = (__bf16)bReg[p].y;
                h[2] = (__bf16)bReg[p].z; h[3] = (__bf16)bReg[p].w;
                *(bf16x4*)&sB[buf][p * 32 + arow][akc] = h;
            }
        } else {                                  // transpose on store
#pragma unroll
            for (int p = 0; p < 4; ++p) {
                int kr = p * 8 + bkr;
                sB[buf][bnc + 0][kr] = (__bf16)bReg[p].x;
                sB[buf][bnc + 1][kr] = (__bf16)bReg[p].y;
                sB[buf][bnc + 2][kr] = (__bf16)bReg[p].z;
                sB[buf][bnc + 3][kr] = (__bf16)bReg[p].w;
            }
        }
        __syncthreads();

        // ---- stage NEXT k-tile while this one computes -----------------
        const int kn = k0 + 32;
        if (kn < K) {
#pragma unroll
            for (int p = 0; p < 4; ++p)
                aReg[p] = *(const float4*)(Ag + (long long)(m0 + p * 32 + arow) * lda + kn + akc);
            if (!bNormal) {
#pragma unroll
                for (int p = 0; p < 4; ++p) {
                    int gn = n0 + p * 32 + arow;
                    bReg[p] = (nfull || gn < N)
                            ? *(const float4*)(Bg + (long long)gn * ldb + kn + akc)
                            : make_float4(0.f, 0.f, 0.f, 0.f);
                }
            } else {
#pragma unroll
                for (int p = 0; p < 4; ++p) {
                    int gk = kn + p * 8 + bkr;
                    bReg[p] = bn_ok
                            ? *(const float4*)(Bg + (long long)gk * ldb + n0 + bnc)
                            : make_float4(0.f, 0.f, 0.f, 0.f);
                }
            }
            if (kn + 32 < K)   // speculative prefetch of tile after next
                __builtin_prefetch(Ag + (long long)(m0 + arow) * lda + kn + 32, 0, 1);
        }

        // ---- fragments per CDNA5 16-bit WMMA layouts -------------------
        const int lr    = lane & 15;
        const int ahalf = (lane >> 4) * 8;    // A: lanes 0-15 K{0..7,16..23}; 16-31 K{8..15,24..31}
        const int bbase = (lane >> 4) * 8;    // B: lanes 0-15 K{0..7,16..23}; 16-31 K{8..15,24..31}

        bf16x16 af[2], bfz[4];
#pragma unroll
        for (int i = 0; i < 2; ++i) {
            int r = wm + i * 16 + lr;
            bf16x8 lo = *(const bf16x8*)&sA[buf][r][ahalf];
            bf16x8 hi = *(const bf16x8*)&sA[buf][r][ahalf + 16];
#pragma unroll
            for (int e = 0; e < 8; ++e) { af[i][e] = lo[e]; af[i][8 + e] = hi[e]; }
        }
#pragma unroll
        for (int j = 0; j < 4; ++j) {
            int c = wn + j * 16 + lr;
            bf16x8 lo = *(const bf16x8*)&sB[buf][c][bbase];
            bf16x8 hi = *(const bf16x8*)&sB[buf][c][bbase + 16];
#pragma unroll
            for (int e = 0; e < 8; ++e) { bfz[j][e] = lo[e]; bfz[j][8 + e] = hi[e]; }
        }
#pragma unroll
        for (int i = 0; i < 2; ++i)
#pragma unroll
            for (int j = 0; j < 4; ++j)
                acc[i][j] = __builtin_amdgcn_wmma_f32_16x16x32_bf16(
                    false, af[i], false, bfz[j], (short)0, acc[i][j], false, false);

        buf ^= 1;
    }

    // ---- epilogue: C/D layout — VGPR e: M=e (lanes 0-15) / M=e+8 (16-31)
    const int lr = lane & 15;
    const int mh = (lane >> 4) * 8;
#pragma unroll
    for (int i = 0; i < 2; ++i) {
#pragma unroll
        for (int j = 0; j < 4; ++j) {
            int n = n0 + wn + j * 16 + lr;
            if (n >= N) continue;
            float bv = bias ? bias[n] : 0.0f;
#pragma unroll
            for (int e = 0; e < 8; ++e) {
                int m = m0 + wm + i * 16 + mh + e;
                float v = acc[i][j][e] * alpha + bv;
                if      (act == 1) v = fmaxf(v, 0.0f);
                else if (act == 2) v = v * fast_sigmoid(v);
                else if (act == 3) v = log1pf(__expf(v));
                if (res) v += res[(long long)m * ldr + n];
                Cg[(long long)m * ldc + n] = v;
            }
        }
    }
}

// ---------------------------------------------------------------------------
// Causal depthwise conv (kernel 4) + bias + SiLU over xr = xz[..., :kDI]
// ---------------------------------------------------------------------------
__global__ __launch_bounds__(256)
void conv_silu_kernel(const float* __restrict__ xz, const float* __restrict__ w,
                      const float* __restrict__ bias, float* __restrict__ xc, int total)
{
    int idx = blockIdx.x * 256 + threadIdx.x;
    if (idx >= total) return;
    int d  = idx % kDI;
    int bl = idx / kDI;
    int l  = bl % kLL;
    int b  = bl / kLL;
    float acc = bias[d];
#pragma unroll
    for (int k = 0; k < 4; ++k) {
        int ls = l - 3 + k;
        if (ls >= 0)
            acc += xz[((long long)(b * kLL + ls)) * (2 * kDI) + d] * w[d * 4 + k];
    }
    xc[idx] = acc * fast_sigmoid(acc);
}

// ---------------------------------------------------------------------------
// Fused selective scan: per (b, channel) sequential over L, 16 states in
// registers; fuses +D*xc skip and *silu(z) gate.
// ---------------------------------------------------------------------------
__global__ __launch_bounds__(256)
void mamba_scan_kernel(const float* __restrict__ dt, const float* __restrict__ xc,
                       const float* __restrict__ xdbl, const float* __restrict__ A_log,
                       const float* __restrict__ Dp, const float* __restrict__ xz,
                       float* __restrict__ yg)
{
    int idx = blockIdx.x * 256 + threadIdx.x;     // kBB * kDI threads
    if (idx >= kBB * kDI) return;
    int d = idx % kDI;
    int b = idx / kDI;
    float Arow[kDS], h[kDS];
#pragma unroll
    for (int s = 0; s < kDS; ++s) { Arow[s] = -__expf(A_log[d * kDS + s]); h[s] = 0.0f; }
    float Dd = Dp[d];
    for (int l = 0; l < kLL; ++l) {
        long long rl = (long long)(b * kLL + l);
        float dtv = dt[rl * kDI + d];
        float xv  = xc[rl * kDI + d];
        const float* xd = xdbl + rl * 96;     // [dt_r(64) | B(16) | C(16)]
        float dx = dtv * xv;
        float y = 0.0f;
#pragma unroll
        for (int s = 0; s < kDS; ++s) {
            float da = __expf(dtv * Arow[s]);
            h[s] = da * h[s] + dx * xd[kDR + s];
            y += h[s] * xd[kDR + kDS + s];
        }
        float zv = xz[rl * (2 * kDI) + kDI + d];
        yg[rl * kDI + d] = (y + Dd * xv) * (zv * fast_sigmoid(zv));
    }
}

// ---------------------------------------------------------------------------
// LayerNorm. custom=0: g*(x-m)*rsqrt(var+eps)+b  (biased var, eps=1e-5)
//            custom=1: g*(x-m)/(std+eps)+b       (ddof=1 std, eps=1e-6)
// one block (256 threads) per row
// ---------------------------------------------------------------------------
__global__ __launch_bounds__(256)
void layernorm_kernel(const float* __restrict__ x, const float* __restrict__ g,
                      const float* __restrict__ bb, float* __restrict__ y,
                      int D, float eps, int custom)
{
    __shared__ float s1[256], s2[256];
    long long row = blockIdx.x;
    const float* xr = x + row * D;
    float sum = 0.f, sq = 0.f;
    for (int i = threadIdx.x; i < D; i += 256) { float v = xr[i]; sum += v; sq += v * v; }
    s1[threadIdx.x] = sum; s2[threadIdx.x] = sq;
    __syncthreads();
    for (int st = 128; st > 0; st >>= 1) {
        if (threadIdx.x < st) {
            s1[threadIdx.x] += s1[threadIdx.x + st];
            s2[threadIdx.x] += s2[threadIdx.x + st];
        }
        __syncthreads();
    }
    float m = s1[0] / D;
    float inv;
    if (custom) {
        float var = (s2[0] - (float)D * m * m) / (float)(D - 1);
        inv = 1.0f / (sqrtf(fmaxf(var, 0.0f)) + eps);
    } else {
        float var = s2[0] / D - m * m;
        inv = rsqrtf(fmaxf(var, 0.0f) + eps);
    }
    float* yr = y + row * D;
    for (int i = threadIdx.x; i < D; i += 256)
        yr[i] = g[i] * (xr[i] - m) * inv + bb[i];
}

// ---------------------------------------------------------------------------
// In-place softmax over rows of 512 (attention scores). One block per row.
// ---------------------------------------------------------------------------
__global__ __launch_bounds__(256)
void softmax512_kernel(float* __restrict__ S)
{
    __shared__ float red[256];
    long long row = blockIdx.x;
    float* r = S + row * 512;
    int t = threadIdx.x;
    float a = r[t], b = r[t + 256];
    red[t] = fmaxf(a, b);
    __syncthreads();
    for (int st = 128; st > 0; st >>= 1) {
        if (t < st) red[t] = fmaxf(red[t], red[t + st]);
        __syncthreads();
    }
    float mx = red[0];
    __syncthreads();
    float e0 = __expf(a - mx), e1 = __expf(b - mx);
    red[t] = e0 + e1;
    __syncthreads();
    for (int st = 128; st > 0; st >>= 1) {
        if (t < st) red[t] += red[t + st];
        __syncthreads();
    }
    float inv = 1.0f / red[0];
    r[t] = e0 * inv;
    r[t + 256] = e1 * inv;
}

// ---------------------------------------------------------------------------
// Host-side orchestration
// ---------------------------------------------------------------------------
struct MambaP { const float *in_w,*conv_w,*conv_b,*x_w,*dt_w,*dt_b,*A_log,*D,*out_w,*ln_g,*ln_b; };
struct MhaP   { const float *sub_g,*sub_b,*wq,*bq,*wk,*bk,*wv,*bv,*wo,*bo; };
struct FfnP   { const float *w1,*b1,*w2,*b2,*g,*b; };
struct Ws     { float *xz,*xc,*xd,*dt,*yg,*Q,*K,*V,*S,*tA,*tB; };

static void launch_gemm(hipStream_t s,
    const float* A, int lda, long long aso, long long asi,
    const float* B, int ldb, long long bso, long long bsi, int bNormal,
    const float* bias, const float* res, int ldr, long long rso, long long rsi,
    float* C, int ldc, long long cso, long long csi,
    int M, int N, int K, int nbatch, int binner, float alpha, int act)
{
    dim3 grid((N + 127) / 128, (M + 127) / 128, nbatch);
    gemm_bf16_wmma<<<grid, 256, 0, s>>>(A, lda, aso, asi, B, ldb, bso, bsi, bNormal,
                                        bias, res, ldr, rso, rsi, C, ldc, cso, csi,
                                        M, N, K, binner, alpha, act);
}

static void run_ln(hipStream_t s, const float* x, const float* g, const float* b,
                   float* y, int rows, int D, float eps, int custom)
{
    layernorm_kernel<<<rows, 256, 0, s>>>(x, g, b, y, D, eps, custom);
}

static void mamba_layer(hipStream_t s, const MambaP& p, const float* xin,
                        float* xstream, const Ws& w)
{
    // xz = x @ in_w^T
    launch_gemm(s, xin, kDM,0,0, p.in_w, kDM,0,0, 0, nullptr, nullptr,0,0,0,
                w.xz, 2*kDI,0,0, kM, 2*kDI, kDM, 1,1, 1.0f, 0);
    // xc = silu(causal_conv(xr) + conv_b)
    int total = kM * kDI;
    conv_silu_kernel<<<(total + 255) / 256, 256, 0, s>>>(w.xz, p.conv_w, p.conv_b, w.xc, total);
    // xdbl = xc @ x_w^T      (N=96, ragged)
    launch_gemm(s, w.xc, kDI,0,0, p.x_w, kDI,0,0, 0, nullptr, nullptr,0,0,0,
                w.xd, 96,0,0, kM, 96, kDI, 1,1, 1.0f, 0);
    // dt = softplus(dt_r @ dt_w^T + dt_b)   (A = first 64 cols of xdbl)
    launch_gemm(s, w.xd, 96,0,0, p.dt_w, kDR,0,0, 0, p.dt_b, nullptr,0,0,0,
                w.dt, kDI,0,0, kM, kDI, kDR, 1,1, 1.0f, 3);
    // fused selective scan + skip + gate
    mamba_scan_kernel<<<(kBB * kDI + 255) / 256, 256, 0, s>>>(
        w.dt, w.xc, w.xd, p.A_log, p.D, w.xz, w.yg);
    // out = yg @ out_w^T + x  (residual)
    launch_gemm(s, w.yg, kDI,0,0, p.out_w, kDI,0,0, 0, nullptr, xin, kDM,0,0,
                w.tA, kDM,0,0, kM, kDM, kDI, 1,1, 1.0f, 0);
    // stream = LN(out)
    run_ln(s, w.tA, p.ln_g, p.ln_b, xstream, kM, kDM, 1e-5f, 0);
}

static void attn_sublayer(hipStream_t s, const MhaP& p, float* srb1, const float* srb2,
                          const float* norm_g, const float* norm_b, const Ws& w)
{
    run_ln(s, srb1, p.sub_g, p.sub_b, w.tA, kM, kDM, 1e-5f, 0);    // LN(q src)
    run_ln(s, srb2, p.sub_g, p.sub_b, w.tB, kM, kDM, 1e-5f, 0);    // n2 = LN(kv src)
    // Q,K,V projections
    launch_gemm(s, w.tA, kDM,0,0, p.wq, kDM,0,0, 0, p.bq, nullptr,0,0,0,
                w.Q, kDM,0,0, kM, kDM, kDM, 1,1, 1.0f, 0);
    launch_gemm(s, w.tB, kDM,0,0, p.wk, kDM,0,0, 0, p.bk, nullptr,0,0,0,
                w.K, kDM,0,0, kM, kDM, kDM, 1,1, 1.0f, 0);
    launch_gemm(s, w.tB, kDM,0,0, p.wv, kDM,0,0, 0, p.bv, nullptr,0,0,0,
                w.V, kDM,0,0, kM, kDM, kDM, 1,1, 1.0f, 0);
    // S = (Q K^T) / sqrt(hd)   per (b,h): M=N=512, K=64
    launch_gemm(s, w.Q, kDM, (long long)kLL*kDM, kHD,
                   w.K, kDM, (long long)kLL*kDM, kHD, 0,
                nullptr, nullptr,0,0,0,
                w.S, kLL, (long long)kNH*kLL*kLL, (long long)kLL*kLL,
                kLL, kLL, kHD, kBB*kNH, kNH, 0.125f, 0);
    softmax512_kernel<<<kBB * kNH * kLL, 256, 0, s>>>(w.S);
    // O = P V   per (b,h): M=512, N=64 (ragged), K=512 (B in [K,N] layout)
    launch_gemm(s, w.S, kLL, (long long)kNH*kLL*kLL, (long long)kLL*kLL,
                   w.V, kDM, (long long)kLL*kDM, kHD, 1,
                nullptr, nullptr,0,0,0,
                w.tA, kDM, (long long)kLL*kDM, kHD,
                kLL, kHD, kLL, kBB*kNH, kNH, 1.0f, 0);
    // out-proj + residual srb1, then outer N(...)
    launch_gemm(s, w.tA, kDM,0,0, p.wo, kDM,0,0, 0, p.bo, srb1, kDM,0,0,
                w.tB, kDM,0,0, kM, kDM, kDM, 1,1, 1.0f, 0);
    run_ln(s, w.tB, norm_g, norm_b, srb1, kM, kDM, 1e-5f, 0);
}

static void ffn_layer(hipStream_t s, const FfnP& p, float* stream, const Ws& w)
{
    launch_gemm(s, stream, kDM,0,0, p.w1, kDM,0,0, 0, p.b1, nullptr,0,0,0,
                w.xc, 2*kDM,0,0, kM, 2*kDM, kDM, 1,1, 1.0f, 1);          // relu
    launch_gemm(s, w.xc, 2*kDM,0,0, p.w2, 2*kDM,0,0, 0, p.b2, stream, kDM,0,0,
                w.tA, kDM,0,0, kM, kDM, 2*kDM, 1,1, 1.0f, 0);
    run_ln(s, w.tA, p.g, p.b, stream, kM, kDM, 1e-6f, 1);                // custom_ln
}

extern "C" void kernel_launch(void* const* d_in, const int* in_sizes, int n_in,
                              void* d_out, int out_size, void* d_ws, size_t ws_size,
                              hipStream_t stream)
{
    (void)in_sizes; (void)n_in; (void)out_size; (void)ws_size;
    auto F = [&](int i) { return (const float*)d_in[i]; };

    const float* text_in  = F(0);
    const float* video_in = F(1);
    const float* audio_in = F(2);

    int idx = 3;
    MambaP mp[6];
    for (int i = 0; i < 6; ++i) {
        mp[i] = { F(idx+0),F(idx+1),F(idx+2),F(idx+3),F(idx+4),F(idx+5),
                  F(idx+6),F(idx+7),F(idx+8),F(idx+9),F(idx+10) };
        idx += 11;
    }
    MhaP hp[9];
    for (int i = 0; i < 9; ++i) {
        hp[i] = { F(idx+0),F(idx+1),F(idx+2),F(idx+3),F(idx+4),
                  F(idx+5),F(idx+6),F(idx+7),F(idx+8),F(idx+9) };
        idx += 10;
    }
    FfnP fp[3];
    for (int i = 0; i < 3; ++i) {
        fp[i] = { F(idx+0),F(idx+1),F(idx+2),F(idx+3),F(idx+4),F(idx+5) };
        idx += 6;
    }
    const float* norm_g = F(idx++);
    const float* norm_b = F(idx++);

    // workspace carve-out (floats)
    float* wsb = (float*)d_ws;
    Ws w;
    size_t o = 0;
    w.xz = wsb + o; o += (size_t)kM * 2 * kDI;
    w.xc = wsb + o; o += (size_t)kM * kDI;
    w.xd = wsb + o; o += (size_t)kM * 96;
    w.dt = wsb + o; o += (size_t)kM * kDI;
    w.yg = wsb + o; o += (size_t)kM * kDI;
    w.Q  = wsb + o; o += (size_t)kM * kDM;
    w.K  = wsb + o; o += (size_t)kM * kDM;
    w.V  = wsb + o; o += (size_t)kM * kDM;
    w.S  = wsb + o; o += (size_t)kBB * kNH * kLL * kLL;
    w.tA = wsb + o; o += (size_t)kM * kDM;
    w.tB = wsb + o; o += (size_t)kM * kDM;

    // streams live directly in d_out (final values end up in place)
    float* T = (float*)d_out;
    float* V = T + (size_t)kM * kDM;
    float* A = V + (size_t)kM * kDM;

    // --- forward pass (sequential cross-modal dependencies respected) ---
    mamba_layer(stream, mp[0], text_in,  T, w);
    mamba_layer(stream, mp[1], video_in, V, w);
    mamba_layer(stream, mp[2], audio_in, A, w);

    attn_sublayer(stream, hp[0], T, T, norm_g, norm_b, w);   // self1
    attn_sublayer(stream, hp[1], V, V, norm_g, norm_b, w);
    attn_sublayer(stream, hp[2], A, A, norm_g, norm_b, w);

    attn_sublayer(stream, hp[3], T, V, norm_g, norm_b, w);   // text  <- video
    attn_sublayer(stream, hp[4], V, T, norm_g, norm_b, w);   // video <- updated text
    attn_sublayer(stream, hp[5], A, T, norm_g, norm_b, w);   // audio <- updated text

    mamba_layer(stream, mp[3], T, T, w);
    mamba_layer(stream, mp[4], V, V, w);
    mamba_layer(stream, mp[5], A, A, w);

    attn_sublayer(stream, hp[6], T, T, norm_g, norm_b, w);   // self2
    attn_sublayer(stream, hp[7], V, V, norm_g, norm_b, w);
    attn_sublayer(stream, hp[8], A, A, norm_g, norm_b, w);

    ffn_layer(stream, fp[0], T, w);
    ffn_layer(stream, fp[1], V, w);
    ffn_layer(stream, fp[2], A, w);
}
